// HourglassNeck_49709951484651
// MI455X (gfx1250) — matrix operand order repaired and checked by reading, not compiled, not run on hardware
//
#include <hip/hip_runtime.h>

// ---------------------------------------------------------------------------
// HourglassNeck on gfx1250 (MI455X, wave32, WMMA bf16 -> f32 accumulate)
// - fragments: 2x ds_load_b128 each (A row-major, B transposed [n][k])
// - GEMM: 64x64 block tile, 32x32 per wave (4 wmma / k-step / wave)
// - float4-vectorized staging + coalesced float4 epilogue via LDS bounce
// ---------------------------------------------------------------------------

typedef __attribute__((ext_vector_type(16))) __bf16 v16bf;
typedef __attribute__((ext_vector_type(8)))  __bf16 v8bf;
typedef __attribute__((ext_vector_type(4)))  __bf16 v4bf;
typedef __attribute__((ext_vector_type(8)))  float  v8f;

#define NB    4      // batch
#define CIN   1536
#define NPIX  576    // 24*24
#define MIDC  768
#define DIMC  384
#define NHEAD 8
#define DHEAD 48
#define NGRP  8
#define GC    48     // channels per offset group
#define NKV   64     // 8*8 sampled keys
#define CPBD  96

// ---- WMMA fragment helpers (ISA 7.12.2 layouts) ---------------------------
// A (16x32 bf16) from row-major [m][ld]: lane m=lane&15, half=lane>>4.
// k = e + 16*half  -> one contiguous 16-elem run (two 8-elem loads).
__device__ inline v16bf frag_a(const __bf16* base, int ld, int lane) {
  const int m = lane & 15, half = lane >> 4;
  const v8bf lo = *(const v8bf*)(base + m * ld + 16 * half);
  const v8bf hi = *(const v8bf*)(base + m * ld + 16 * half + 8);
  return __builtin_shufflevector(lo, hi, 0, 1, 2, 3, 4, 5, 6, 7,
                                 8, 9, 10, 11, 12, 13, 14, 15);
}
// B (32x16 bf16) from TRANSPOSED [n][ld] storage: lane n=lane&15, k=e+16*half.
__device__ inline v16bf frag_bt(const __bf16* base, int ld, int lane) {
  const int n = lane & 15, half = lane >> 4;
  const v8bf lo = *(const v8bf*)(base + n * ld + 16 * half);
  const v8bf hi = *(const v8bf*)(base + n * ld + 16 * half + 8);
  return __builtin_shufflevector(lo, hi, 0, 1, 2, 3, 4, 5, 6, 7,
                                 8, 9, 10, 11, 12, 13, 14, 15);
}
__device__ inline v8f wmma_bf16(v16bf a, v16bf b, v8f c) {
  return __builtin_amdgcn_wmma_f32_16x16x32_bf16(false, a, false, b, (short)0, c,
                                                 false, false);
}

// ---------------------------------------------------------------------------
// conv1x1-as-GEMM: Out[b][m][p] = relu(sum_k W[m][k]*X[b][k][p] [+bias] [+res])
// Block = 128 threads / 4 waves in a 2x2 grid; wave owns a 32x32 C strip.
// ---------------------------------------------------------------------------
template <bool HAS_BIAS, bool HAS_RES>
__global__ __launch_bounds__(128) void gemm_kernel(
    const float* __restrict__ Wm, const float* __restrict__ X,
    const float* __restrict__ bias, const float* __restrict__ resid,
    float* __restrict__ Out, int M, int K, int N) {
  __shared__ __align__(16) __bf16 Ws[64][40];   // [m][k], padded stride
  __shared__ __align__(16) __bf16 XsT[64][40];  // [n][k] transposed, padded
  __shared__ __align__(16) float  Cs[64][68];   // epilogue bounce
  const int tid  = threadIdx.x;
  const int lane = tid & 31;
  const int wave = tid >> 5;
  const int wm = wave & 1, wn = wave >> 1;
  const int n0 = blockIdx.x * 64;
  const int m0 = blockIdx.y * 64;
  const int b  = blockIdx.z;
  const float* Xb = X + (size_t)b * K * N;

  const v8f z = {};
  v8f acc00 = z, acc01 = z, acc10 = z, acc11 = z;
  for (int k0 = 0; k0 < K; k0 += 32) {
    // W tile: 64x32, float4 loads -> packed bf16x4 LDS stores
    for (int t = tid; t < 512; t += 128) {
      const int m = t >> 3, k = (t & 7) * 4;
      const float4 w4 = *(const float4*)&Wm[(size_t)(m0 + m) * K + (k0 + k)];
      const v4bf p = {(__bf16)w4.x, (__bf16)w4.y, (__bf16)w4.z, (__bf16)w4.w};
      *(v4bf*)&Ws[m][k] = p;
    }
    // X tile: 32k x 64n, float4 along n (coalesced), scatter-transpose to LDS
    for (int t = tid; t < 512; t += 128) {
      const int k = t >> 4, n = (t & 15) * 4;
      const float4 x4 = *(const float4*)&Xb[(size_t)(k0 + k) * N + (n0 + n)];
      XsT[n + 0][k] = (__bf16)x4.x;
      XsT[n + 1][k] = (__bf16)x4.y;
      XsT[n + 2][k] = (__bf16)x4.z;
      XsT[n + 3][k] = (__bf16)x4.w;
    }
    __syncthreads();
    const v16bf a0 = frag_a(&Ws[wm * 32][0], 40, lane);
    const v16bf a1 = frag_a(&Ws[wm * 32 + 16][0], 40, lane);
    const v16bf b0 = frag_bt(&XsT[wn * 32][0], 40, lane);
    const v16bf b1 = frag_bt(&XsT[wn * 32 + 16][0], 40, lane);
    acc00 = wmma_bf16(a0, b0, acc00);
    acc01 = wmma_bf16(a0, b1, acc01);
    acc10 = wmma_bf16(a1, b0, acc10);
    acc11 = wmma_bf16(a1, b1, acc11);
    __syncthreads();
  }
  // scatter C tiles to LDS (lane: col = lane&15, rows g + 8*(lane>>4))
  const int cn = lane & 15, cm = 8 * (lane >> 4);
#pragma unroll
  for (int g = 0; g < 8; ++g) {
    Cs[wm * 32 + cm + g][wn * 32 + cn]           = acc00[g];
    Cs[wm * 32 + cm + g][wn * 32 + 16 + cn]      = acc01[g];
    Cs[wm * 32 + 16 + cm + g][wn * 32 + cn]      = acc10[g];
    Cs[wm * 32 + 16 + cm + g][wn * 32 + 16 + cn] = acc11[g];
  }
  __syncthreads();
  // coalesced float4 epilogue
  for (int t = tid; t < 1024; t += 128) {
    const int m = t >> 4, n = (t & 15) * 4;
    const float4 c4 = *(const float4*)&Cs[m][n];
    float vx = c4.x, vy = c4.y, vz = c4.z, vw = c4.w;
    const int gm = m0 + m;
    if (HAS_BIAS) {
      const float bv = bias[gm];
      vx += bv; vy += bv; vz += bv; vw += bv;
    }
    const size_t idx = ((size_t)b * M + gm) * N + (n0 + n);
    if (HAS_RES) {
      const float4 r4 = *(const float4*)&resid[idx];
      vx += r4.x; vy += r4.y; vz += r4.z; vw += r4.w;
    }
    float4 o4;
    o4.x = fmaxf(vx, 0.f); o4.y = fmaxf(vy, 0.f);
    o4.z = fmaxf(vz, 0.f); o4.w = fmaxf(vw, 0.f);
    *(float4*)&Out[idx] = o4;
  }
}

// ---------------------------------------------------------------------------
// Grouped 1x1 conv: out[b][g*48+o][p] = sum_i Wg[g][o][i] * X[b][g*48+i][p]
// ---------------------------------------------------------------------------
__global__ void grouped_conv_kernel(const float* __restrict__ Wg,
                                    const float* __restrict__ X,
                                    float* __restrict__ Out, int N) {
  const int idx = blockIdx.x * blockDim.x + threadIdx.x;
  const int total = NB * DIMC * N;
  if (idx >= total) return;
  const int p = idx % N;
  const int c = (idx / N) % DIMC;
  const int b = idx / (N * DIMC);
  const int g = c / GC, o = c % GC;
  const float* w   = Wg + (size_t)(g * GC + o) * GC;
  const float* xin = X + ((size_t)b * DIMC + g * GC) * N + p;
  float acc = 0.f;
#pragma unroll 8
  for (int i = 0; i < GC; ++i) acc += w[i] * xin[(size_t)i * N];
  Out[((size_t)b * DIMC + c) * N + p] = acc;
}

// ---------------------------------------------------------------------------
// Offsets: depthwise 3x3 stride-3 + bias -> exact GELU -> 1x1 (48->2) -> tanh*4
// then normalized sampling grid vgrid[bg][py*8+px][{x,y}]
// ---------------------------------------------------------------------------
__global__ void offsets_kernel(const float* __restrict__ q,
                               const float* __restrict__ Wdw,
                               const float* __restrict__ bdw,
                               const float* __restrict__ Wpw,
                               float* __restrict__ vgrid) {
  const int idx = blockIdx.x * blockDim.x + threadIdx.x;
  if (idx >= 32 * 64) return;
  const int px = idx % 8, py = (idx / 8) % 8, bg = idx / 64;
  const int b = bg / NGRP, g = bg % NGRP;
  float acc0 = 0.f, acc1 = 0.f;
  for (int c = 0; c < GC; ++c) {
    const float* img = q + ((size_t)b * DIMC + g * GC + c) * NPIX;
    float a = bdw[c];
#pragma unroll
    for (int ky = 0; ky < 3; ++ky)
#pragma unroll
      for (int kx = 0; kx < 3; ++kx)
        a += Wdw[c * 9 + ky * 3 + kx] * img[(py * 3 + ky) * 24 + (px * 3 + kx)];
    const float ge = 0.5f * a * (1.f + erff(a * 0.70710678118654752f));
    acc0 += ge * Wpw[c];        // Wpw[0][c]
    acc1 += ge * Wpw[GC + c];   // Wpw[1][c]
  }
  const float off0 = tanhf(acc0) * 4.0f;
  const float off1 = tanhf(acc1) * 4.0f;
  vgrid[idx * 2 + 0] = 2.f * ((float)px + off0) / 7.f - 1.f;
  vgrid[idx * 2 + 1] = 2.f * ((float)py + off1) / 7.f - 1.f;
}

// ---------------------------------------------------------------------------
// Bilinear grid sample (zeros padding, align_corners=False) on 24x24 images.
// ---------------------------------------------------------------------------
__global__ void grid_sample_kernel(const float* __restrict__ h2,
                                   const float* __restrict__ vgrid,
                                   float* __restrict__ kv) {
  const int idx = blockIdx.x * blockDim.x + threadIdx.x;
  if (idx >= 32 * GC * NKV) return;
  const int p  = idx % NKV;
  const int c  = (idx / NKV) % GC;
  const int bg = idx / (NKV * GC);
  const int b = bg / NGRP, g = bg % NGRP;
  const float gx = vgrid[(bg * NKV + p) * 2 + 0];
  const float gy = vgrid[(bg * NKV + p) * 2 + 1];
  const float x = ((gx + 1.f) * 24.f - 1.f) * 0.5f;
  const float y = ((gy + 1.f) * 24.f - 1.f) * 0.5f;
  const float x0f = floorf(x), y0f = floorf(y);
  const float wx = x - x0f, wy = y - y0f;
  const int x0 = (int)x0f, y0 = (int)y0f;
  const float* img = h2 + ((size_t)b * DIMC + g * GC + c) * NPIX;
  float v = 0.f;
#pragma unroll
  for (int t = 0; t < 4; ++t) {
    const int ix = x0 + (t & 1), iy = y0 + (t >> 1);
    const float wgt = ((t & 1) ? wx : 1.f - wx) * ((t >> 1) ? wy : 1.f - wy);
    if (ix >= 0 && ix < 24 && iy >= 0 && iy < 24) v += img[iy * 24 + ix] * wgt;
  }
  kv[((size_t)b * DIMC + g * GC + c) * NKV + p] = v;
}

// ---------------------------------------------------------------------------
// Fused CPB MLP: per row (bg, i, j): pos -> sign*log1p -> 2->96 relu (VALU)
//   -> 96x96 relu via WMMA -> dot with cw3 + cb3. 16 rows per wave.
// ---------------------------------------------------------------------------
__global__ __launch_bounds__(128) void cpb_kernel(
    const float* __restrict__ vgrid, const float* __restrict__ cw1,
    const float* __restrict__ cb1, const float* __restrict__ cw2,
    const float* __restrict__ cb2, const float* __restrict__ cw3,
    const float* __restrict__ cb3, float* __restrict__ biasb) {
  __shared__ __align__(16) __bf16 W2nk[CPBD][104];  // [n][k] == cw2 row-major
  __shared__ __align__(16) __bf16 A1[4][16][104];   // act1 [r][k], padded
  __shared__ float Red[4][16][CPBD];                // layer-3 partials
  const int tid = threadIdx.x, lane = tid & 31, wave = tid >> 5;
  const int bg   = blockIdx.y;
  const int row0 = (blockIdx.x * 4 + wave) * 16;

  for (int t = tid; t < (CPBD * CPBD) / 4; t += 128) {
    const int l = t * 4, n = l / CPBD, k = l % CPBD;  // 96%4==0 -> same row
    const float4 w4 = *(const float4*)&cw2[l];
    const v4bf p = {(__bf16)w4.x, (__bf16)w4.y, (__bf16)w4.z, (__bf16)w4.w};
    *(v4bf*)&W2nk[n][k] = p;
  }
  for (int t = lane; t < 16 * CPBD; t += 32) {
    const int r = t / CPBD, kc = t % CPBD;
    const int row = row0 + r;
    const int i = row >> 6, j = row & 63;
    const float q0 = 2.f * (float)(i % 24) / 23.f - 1.f;
    const float q1 = 2.f * (float)(i / 24) / 23.f - 1.f;
    const float p0 = q0 - vgrid[(bg * NKV + j) * 2 + 0];
    const float p1 = q1 - vgrid[(bg * NKV + j) * 2 + 1];
    const float t0 = copysignf(log1pf(fabsf(p0)), p0);
    const float t1 = copysignf(log1pf(fabsf(p1)), p1);
    const float a = t0 * cw1[kc * 2 + 0] + t1 * cw1[kc * 2 + 1] + cb1[kc];
    A1[wave][r][kc] = (__bf16)fmaxf(a, 0.f);
  }
  __syncthreads();

  v8f acc[6];
  const v8f z = {};
#pragma unroll
  for (int i = 0; i < 6; ++i) acc[i] = z;
#pragma unroll
  for (int ks = 0; ks < 3; ++ks) {
    const v16bf a = frag_a(&A1[wave][0][ks * 32], 104, lane);
#pragma unroll
    for (int nt = 0; nt < 6; ++nt) {
      const v16bf bb = frag_bt(&W2nk[nt * 16][ks * 32], 104, lane);
      acc[nt] = wmma_bf16(a, bb, acc[nt]);
    }
  }
#pragma unroll
  for (int nt = 0; nt < 6; ++nt) {
    const int n = nt * 16 + (lane & 15);
    const float w3 = cw3[n], b2 = cb2[n];
#pragma unroll
    for (int g = 0; g < 8; ++g) {
      const int m = g + 8 * (lane >> 4);
      Red[wave][m][n] = fmaxf(acc[nt][g] + b2, 0.f) * w3;
    }
  }
  __syncthreads();
  if (lane < 16) {
    float s = cb3[0];
    for (int n = 0; n < CPBD; ++n) s += Red[wave][lane][n];
    biasb[(size_t)bg * (NPIX * NKV) + row0 + lane] = s;
  }
}

// ---------------------------------------------------------------------------
// Fused attention per (b, head, 16-query tile): sim WMMA (K=48 pad 64),
// +bias, row softmax via LDS, attn @ V WMMA (K=64).
// ---------------------------------------------------------------------------
__global__ __launch_bounds__(128) void attention_kernel(
    const float* __restrict__ qb, const float* __restrict__ kb,
    const float* __restrict__ vb, const float* __restrict__ biasb,
    float* __restrict__ outb) {
  __shared__ __align__(16) __bf16 ksT[64][72];     // [j][d], d>=48 zeroed
  __shared__ __align__(16) __bf16 vsT[DHEAD][72];  // [d][j]
  __shared__ __align__(16) __bf16 qs[4][16][64];   // scaled q [m][d], padded
  __shared__ float simf[4][16][64];
  __shared__ __align__(16) __bf16 pbf[4][16][64];  // probs [m][j]
  const int tid = threadIdx.x, lane = tid & 31, wave = tid >> 5;
  const int h = blockIdx.y, b = blockIdx.z;
  const int i0 = (blockIdx.x * 4 + wave) * 16;
  const float scale = 0.14433756729740643f;  // 48^-0.5

  for (int t = tid; t < 64 * 64; t += 128) {
    const int j = t & 63, d = t >> 6;  // j fast -> coalesced global reads
    ksT[j][d] = (d < DHEAD)
                    ? (__bf16)kb[((size_t)b * DIMC + h * DHEAD + d) * NKV + j]
                    : (__bf16)0.f;
  }
  for (int t = tid; t < DHEAD * 64; t += 128) {
    const int j = t & 63, d = t >> 6;
    vsT[d][j] = (__bf16)vb[((size_t)b * DIMC + h * DHEAD + d) * NKV + j];
  }
  for (int t = tid; t < 4 * 16 * 64; t += 128) {
    const int d = t & 63, m = (t >> 6) & 15, w = t >> 10;
    const int i = blockIdx.x * 64 + w * 16 + m;
    qs[w][m][d] =
        (d < DHEAD)
            ? (__bf16)(qb[((size_t)b * DIMC + h * DHEAD + d) * NPIX + i] * scale)
            : (__bf16)0.f;
  }
  __syncthreads();

  const v8f z = {};
  v8f s[4];
#pragma unroll
  for (int i = 0; i < 4; ++i) s[i] = z;
#pragma unroll
  for (int ks = 0; ks < 2; ++ks) {
    const v16bf a = frag_a(&qs[wave][0][ks * 32], 64, lane);
#pragma unroll
    for (int nt = 0; nt < 4; ++nt) {
      const v16bf bb = frag_bt(&ksT[nt * 16][ks * 32], 72, lane);
      s[nt] = wmma_bf16(a, bb, s[nt]);
    }
  }
  const size_t bg = (size_t)(b * NGRP + h);
#pragma unroll
  for (int nt = 0; nt < 4; ++nt) {
    const int j = nt * 16 + (lane & 15);
#pragma unroll
    for (int g = 0; g < 8; ++g) {
      const int m = g + 8 * (lane >> 4);
      simf[wave][m][j] =
          s[nt][g] + biasb[bg * (NPIX * NKV) + (size_t)(i0 + m) * NKV + j];
    }
  }
  __syncthreads();
  if (lane < 16) {
    float mx = -1e30f;
    for (int j = 0; j < 64; ++j) mx = fmaxf(mx, simf[wave][lane][j]);
    float sum = 0.f;
    for (int j = 0; j < 64; ++j) {
      const float e = expf(simf[wave][lane][j] - mx);
      simf[wave][lane][j] = e;
      sum += e;
    }
    const float r = 1.f / sum;
    for (int j = 0; j < 64; ++j)
      pbf[wave][lane][j] = (__bf16)(simf[wave][lane][j] * r);
  }
  __syncthreads();

  v8f o[3];
#pragma unroll
  for (int i = 0; i < 3; ++i) o[i] = z;
#pragma unroll
  for (int ks = 0; ks < 2; ++ks) {
    const v16bf a = frag_a(&pbf[wave][0][ks * 32], 64, lane);
#pragma unroll
    for (int nt = 0; nt < 3; ++nt) {
      const v16bf bb = frag_bt(&vsT[nt * 16][ks * 32], 72, lane);
      o[nt] = wmma_bf16(a, bb, o[nt]);
    }
  }
#pragma unroll
  for (int nt = 0; nt < 3; ++nt) {
    const int d = nt * 16 + (lane & 15);
#pragma unroll
    for (int g = 0; g < 8; ++g) {
      const int m = g + 8 * (lane >> 4);
      outb[((size_t)b * DIMC + h * DHEAD + d) * NPIX + (i0 + m)] = o[nt][g];
    }
  }
}

// ---------------------------------------------------------------------------
extern "C" void kernel_launch(void* const* d_in, const int* in_sizes, int n_in,
                              void* d_out, int out_size, void* d_ws,
                              size_t ws_size, hipStream_t stream) {
  (void)in_sizes; (void)n_in; (void)out_size; (void)ws_size;
  const float* x    = (const float*)d_in[0];
  const float* W1   = (const float*)d_in[1];
  const float* W2   = (const float*)d_in[2];
  const float* W3   = (const float*)d_in[3];
  const float* W4   = (const float*)d_in[4];
  const float* Wq   = (const float*)d_in[5];
  const float* Wk   = (const float*)d_in[6];
  const float* Wv   = (const float*)d_in[7];
  const float* Wdw  = (const float*)d_in[8];
  const float* bdw  = (const float*)d_in[9];
  const float* Wpw  = (const float*)d_in[10];
  const float* cw1  = (const float*)d_in[11];
  const float* cb1  = (const float*)d_in[12];
  const float* cw2  = (const float*)d_in[13];
  const float* cb2  = (const float*)d_in[14];
  const float* cw3  = (const float*)d_in[15];
  const float* cb3  = (const float*)d_in[16];
  const float* Wout = (const float*)d_in[17];
  const float* bout = (const float*)d_in[18];
  float* out = (float*)d_out;

  // Workspace carve (~34.3 MB total)
  char* ws = (char*)d_ws;
  size_t off = 0;
  auto alloc = [&](size_t nf) {
    off = (off + 255) & ~(size_t)255;
    void* p = ws + off;
    off += nf * sizeof(float);
    return (float*)p;
  };
  float* h1    = alloc((size_t)NB * MIDC * NPIX);
  float* h2    = alloc((size_t)NB * DIMC * NPIX);
  float* qbuf  = alloc((size_t)NB * DIMC * NPIX);
  float* vgrid = alloc(32 * NKV * 2);
  float* kvb   = alloc((size_t)NB * DIMC * NKV);
  float* kbuf  = alloc((size_t)NB * DIMC * NKV);
  float* vbuf  = alloc((size_t)NB * DIMC * NKV);
  float* biasb = alloc((size_t)32 * NPIX * NKV);
  float* aout  = alloc((size_t)NB * DIMC * NPIX);
  float* proj  = alloc((size_t)NB * DIMC * NPIX);
  float* h3    = alloc((size_t)NB * MIDC * NPIX);

  // L1: 1536 -> 768, relu
  gemm_kernel<false, false><<<dim3(9, MIDC / 64, NB), 128, 0, stream>>>(
      W1, x, nullptr, nullptr, h1, MIDC, CIN, NPIX);
  // L2: 768 -> 384, relu
  gemm_kernel<false, false><<<dim3(9, DIMC / 64, NB), 128, 0, stream>>>(
      W2, h1, nullptr, nullptr, h2, DIMC, MIDC, NPIX);
  // q (grouped 48x48)
  grouped_conv_kernel<<<(NB * DIMC * NPIX + 255) / 256, 256, 0, stream>>>(
      Wq, h2, qbuf, NPIX);
  // offsets -> sampling grid
  offsets_kernel<<<(32 * NKV + 255) / 256, 256, 0, stream>>>(qbuf, Wdw, bdw,
                                                             Wpw, vgrid);
  // bilinear sample kv
  grid_sample_kernel<<<(32 * GC * NKV + 255) / 256, 256, 0, stream>>>(
      h2, vgrid, kvb);
  // k, v (grouped 48x48 on 64 pixels)
  grouped_conv_kernel<<<(NB * DIMC * NKV + 255) / 256, 256, 0, stream>>>(
      Wk, kvb, kbuf, NKV);
  grouped_conv_kernel<<<(NB * DIMC * NKV + 255) / 256, 256, 0, stream>>>(
      Wv, kvb, vbuf, NKV);
  // fused CPB bias MLP (dominant GEMM, WMMA)
  cpb_kernel<<<dim3(576, 32), 128, 0, stream>>>(vgrid, cw1, cb1, cw2, cb2, cw3,
                                                cb3, biasb);
  // fused attention (sim WMMA + softmax + attn@V WMMA)
  attention_kernel<<<dim3(9, NHEAD, NB), 128, 0, stream>>>(qbuf, kbuf, vbuf,
                                                           biasb, aout);
  // output projection + bias, relu (relu from outer wrapper)
  gemm_kernel<true, false><<<dim3(9, DIMC / 64, NB), 128, 0, stream>>>(
      Wout, aout, bout, nullptr, proj, DIMC, DIMC, NPIX);
  // L3: 384 -> 768, relu
  gemm_kernel<false, false><<<dim3(9, MIDC / 64, NB), 128, 0, stream>>>(
      W3, proj, nullptr, nullptr, h3, MIDC, DIMC, NPIX);
  // L4: 768 -> 1536 + residual x, relu
  gemm_kernel<false, true><<<dim3(9, CIN / 64, NB), 128, 0, stream>>>(
      W4, h3, nullptr, x, out, CIN, MIDC, NPIX);
}
